// TransGQARopeSparseMoeBlock_46480136077926
// MI455X (gfx1250) — compile-verified, matrix operand order checked by reading, not run on hardware
//
#include <hip/hip_runtime.h>
#include <hip/hip_bf16.h>
#include <stdint.h>

// ---------------------------------------------------------------- types
typedef float  v8f   __attribute__((ext_vector_type(8)));
typedef __bf16 v16bf __attribute__((ext_vector_type(16)));

constexpr int T = 4096, H = 1024, E = 16, I = 768, TOPK = 8;
constexpr int TM = 32;                 // tokens per expert tile (2 M-tiles of 16)
constexpr int TILES_PER_E = T / TM;    // 128 (worst case: all tokens -> one expert)
constexpr int KBLK_H = H / 32;         // 32 K-blocks for GEMM1
constexpr int KBLK_I = I / 32;         // 24 K-blocks for GEMM2
constexpr int NBLK_I = I / 16;         // 48 N-tiles for GEMM1
constexpr int NBLK_H = H / 16;         // 64 N-tiles for GEMM2

// ---------------------------------------------------------------- helpers
__device__ __forceinline__ unsigned short f2bf(float f) {
  unsigned int u = __builtin_bit_cast(unsigned int, f);
  u += 0x7FFFu + ((u >> 16) & 1u);     // round-to-nearest-even
  return (unsigned short)(u >> 16);
}

struct U32x8 { uint4 lo, hi; };

// A operand (16x32 bf16): lane l: m = l&15, kbase = (l>>4)*8.
// VGPR0..3 hold K = kbase..kbase+7, VGPR4..7 hold K = 16+kbase..16+kbase+7.
__device__ __forceinline__ v16bf loadA(const unsigned short* rowp, int k) {
  U32x8 s;
  s.lo = *(const uint4*)(rowp + k);        // K = k .. k+7
  s.hi = *(const uint4*)(rowp + k + 16);   // K = k+16 .. k+23
  return __builtin_bit_cast(v16bf, s);
}

// B operand from pre-swizzled storage: each 32K x 16N block is 512 bf16 =
// 1 KB; lane l owns 16 contiguous bf16 at blkbase + l*16 (wave-coalesced).
__device__ __forceinline__ v16bf loadB_swz(const unsigned short* blkbase, int lane) {
  const unsigned short* p = blkbase + lane * 16;
  U32x8 s;
  s.lo = *(const uint4*)p;
  s.hi = *(const uint4*)(p + 8);
  return __builtin_bit_cast(v16bf, s);
}

__device__ __forceinline__ v8f wmma_bf16(v16bf a, v16bf b, v8f c) {
  return __builtin_amdgcn_wmma_f32_16x16x32_bf16(false, a, false, b, (short)0, c,
                                                 false, false);
}

// ---------------------------------------------------------------- kernel 1: zero out + counts
__global__ __launch_bounds__(256) void k_zero(float* __restrict__ out, int* __restrict__ counts) {
  int gid = blockIdx.x * blockDim.x + threadIdx.x;
  float4* o4 = (float4*)out;
  if (gid < (T * H) / 4) o4[gid] = make_float4(0.f, 0.f, 0.f, 0.f);
  if (blockIdx.x == 0 && threadIdx.x < E) counts[threadIdx.x] = 0;
}

// ---------------------------------------------------------------- kernel 2: x fp32 -> bf16
__global__ __launch_bounds__(256) void k_cvt_x(const float* __restrict__ x,
                                               unsigned short* __restrict__ xb) {
  int gid = blockIdx.x * blockDim.x + threadIdx.x;
  int base = gid * 4;
  if (base < T * H) {
    float4 v = *(const float4*)(x + base);
    ushort4 r;
    r.x = f2bf(v.x); r.y = f2bf(v.y); r.z = f2bf(v.z); r.w = f2bf(v.w);
    *(ushort4*)(xb + base) = r;
  }
}

// ---------------------------------------------------------------- kernel 3: weight swizzle
// W is [E][Ktot][Ntot] fp32 row-major (contraction dim K = row index).
// Output per expert: [ntile][kblock][512]; within a block lane l (0..31)
// holds its 16 WMMA B-operand bf16 values contiguously:
//   n = ntile*16 + (l&15), kbase = kblock*32 + (l>>4)*8
//   slots 0..7  -> K = kbase .. kbase+7 ; slots 8..15 -> K = kbase+16 .. +23
__global__ __launch_bounds__(256) void k_swz_w(const float* __restrict__ W,
                                               unsigned short* __restrict__ Wb,
                                               int kblkN, int nblkN, int ld) {
  int gid = blockIdx.x * blockDim.x + threadIdx.x;  // one thread = one lane-slot (32B)
  int total = E * kblkN * nblkN * 32;
  if (gid >= total) return;
  int lane = gid & 31;
  int blk  = gid >> 5;
  int perE = kblkN * nblkN;
  int e  = blk / perE;
  int r  = blk - e * perE;
  int nb = r / kblkN;
  int kb = r - nb * kblkN;

  int n     = nb * 16 + (lane & 15);
  int kbase = kb * 32 + (lane >> 4) * 8;
  const float* src = W + (size_t)e * kblkN * 32 * ld;

  unsigned short v[16];
#pragma unroll
  for (int j = 0; j < 8; ++j) v[j]     = f2bf(src[(size_t)(kbase + j) * ld + n]);
#pragma unroll
  for (int j = 0; j < 8; ++j) v[8 + j] = f2bf(src[(size_t)(kbase + 16 + j) * ld + n]);

  unsigned short* dst = Wb + ((size_t)blk * 512) + lane * 16;
  U32x8 s = __builtin_bit_cast(U32x8, v);
  *(uint4*)dst       = s.lo;
  *(uint4*)(dst + 8) = s.hi;
}

// ---------------------------------------------------------------- kernel 4: router (1 wave / token)
__global__ __launch_bounds__(128) void k_router(const float* __restrict__ x,
                                                const float* __restrict__ gate_w,
                                                int* __restrict__ topk_idx,
                                                float* __restrict__ topk_w,
                                                int* __restrict__ counts) {
  int wave = threadIdx.x >> 5, lane = threadIdx.x & 31;
  int t = blockIdx.x * 4 + wave;
  if (t >= T) return;

  float p[E];
#pragma unroll
  for (int e = 0; e < E; ++e) p[e] = 0.f;

  for (int h = lane; h < H; h += 32) {
    float xv = x[(size_t)t * H + h];
    const float* g = gate_w + (size_t)h * E;
#pragma unroll
    for (int e = 0; e < E; ++e) p[e] += xv * g[e];
  }
#pragma unroll
  for (int off = 16; off > 0; off >>= 1) {
#pragma unroll
    for (int e = 0; e < E; ++e) p[e] += __shfl_xor(p[e], off, 32);
  }

  if (lane == 0) {
    bool used[E];
#pragma unroll
    for (int e = 0; e < E; ++e) used[e] = false;
    int ids[TOPK]; float lg[TOPK];
    for (int k = 0; k < TOPK; ++k) {
      float best = -3.4e38f; int bi = 0;
      for (int e = 0; e < E; ++e)
        if (!used[e] && p[e] > best) { best = p[e]; bi = e; }
      used[bi] = true; ids[k] = bi; lg[k] = best;
    }
    float m = lg[0], s = 0.f, w[TOPK];
    for (int k = 0; k < TOPK; ++k) { w[k] = __expf(lg[k] - m); s += w[k]; }
    float inv = 1.f / s;
    for (int k = 0; k < TOPK; ++k) {
      topk_idx[t * TOPK + k] = ids[k];
      topk_w[t * TOPK + k]   = w[k] * inv;
      atomicAdd(&counts[ids[k]], 1);
    }
  }
}

// ---------------------------------------------------------------- kernel 5: 16-entry scan
__global__ void k_scan(const int* __restrict__ counts, int* __restrict__ offsets,
                       int* __restrict__ cursor) {
  if (threadIdx.x == 0) {
    int off = 0;
    for (int e = 0; e < E; ++e) {
      offsets[e] = off; cursor[e] = off; off += counts[e];
    }
  }
}

// ---------------------------------------------------------------- kernel 6: scatter token lists
__global__ __launch_bounds__(256) void k_scatter(const int* __restrict__ topk_idx,
                                                 const float* __restrict__ topk_w,
                                                 int* __restrict__ cursor,
                                                 int* __restrict__ token_list,
                                                 float* __restrict__ weight_list) {
  int t = blockIdx.x * blockDim.x + threadIdx.x;
  if (t >= T) return;
#pragma unroll
  for (int k = 0; k < TOPK; ++k) {
    int e = topk_idx[t * TOPK + k];
    int pos = atomicAdd(&cursor[e], 1);
    token_list[pos]  = t;
    weight_list[pos] = topk_w[t * TOPK + k];
  }
}

// ---------------------------------------------------------------- kernel 7: expert SwiGLU GEMMs
__global__ __launch_bounds__(256) void k_moe_gemm(const unsigned short* __restrict__ xb,
                                                  const unsigned short* __restrict__ w1b,
                                                  const unsigned short* __restrict__ w3b,
                                                  const unsigned short* __restrict__ w2b,
                                                  const int* __restrict__ offsets,
                                                  const int* __restrict__ counts,
                                                  const int* __restrict__ token_list,
                                                  const float* __restrict__ weight_list,
                                                  float* __restrict__ out) {
  const int e    = blockIdx.x >> 7;          // / TILES_PER_E
  const int tile = blockIdx.x & (TILES_PER_E - 1);
  const int cnt  = counts[e];
  const int row0 = tile * TM;
  if (row0 >= cnt) return;
  const int base = offsets[e];

  __shared__ unsigned short s_act[TM * I];   // 48 KB bf16 activation tile
  __shared__ int   s_tok[TM];
  __shared__ float s_wt[TM];

  const int tid = threadIdx.x;
  if (tid < TM) {
    int idx = row0 + tid;
    if (idx < cnt) { s_tok[tid] = token_list[base + idx]; s_wt[tid] = weight_list[base + idx]; }
    else           { s_tok[tid] = token_list[base + row0]; s_wt[tid] = 0.f; }
  }
  __syncthreads();

  const int wv = tid >> 5;          // wave id 0..7
  const int l  = tid & 31;          // lane
  const int n  = l & 15;            // column-in-tile / A row m
  const int kb = (l >> 4) * 8;      // K sub-offset for this lane half
  const int rbase = 8 * (l >> 4);   // C/D: VGPR r holds row r + rbase

  const unsigned short* arow0 = xb + (size_t)s_tok[n]      * H;  // M-tile 0
  const unsigned short* arow1 = xb + (size_t)s_tok[16 + n] * H;  // M-tile 1
  const unsigned short* W1e = w1b + (size_t)e * H * I;   // [nt][ks][512] swizzled
  const unsigned short* W3e = w3b + (size_t)e * H * I;
  const unsigned short* W2e = w2b + (size_t)e * I * H;

  // -------- GEMM1: act[32 x 768] = silu(X*W1) * (X*W3)
  for (int p = 0; p < 2; ++p) {                 // wave covers 6 N-tiles, 3 per pass
    v8f hacc[2][3] = {};
    v8f uacc[2][3] = {};
    const int nt0   = wv * 6 + p * 3;
    const int ncol0 = nt0 * 16 + n;

#pragma unroll 1
    for (int ks = 0; ks < KBLK_H; ++ks) {
      const int k0 = ks * 32 + kb;
      // ---- issue the whole iteration's loads first (one long clause)
      v16bf a0 = loadA(arow0, k0);
      v16bf a1 = loadA(arow1, k0);
      v16bf b1v[3], b3v[3];
#pragma unroll
      for (int j = 0; j < 3; ++j) {
        const unsigned short* b1p = W1e + ((size_t)(nt0 + j) * KBLK_H + ks) * 512;
        const unsigned short* b3p = W3e + ((size_t)(nt0 + j) * KBLK_H + ks) * 512;
        b1v[j] = loadB_swz(b1p, l);
        b3v[j] = loadB_swz(b3p, l);
        // unconditional next-block prefetch: layout is [nt][ks][512]
        // contiguous, so +1KB always lands inside the workspace.
        __builtin_prefetch(b1p + 512, 0, 1);
        __builtin_prefetch(b3p + 512, 0, 1);
      }
      // ---- then the 12 matrix ops
#pragma unroll
      for (int j = 0; j < 3; ++j) {
        hacc[0][j] = wmma_bf16(a0, b1v[j], hacc[0][j]);
        hacc[1][j] = wmma_bf16(a1, b1v[j], hacc[1][j]);
        uacc[0][j] = wmma_bf16(a0, b3v[j], uacc[0][j]);
        uacc[1][j] = wmma_bf16(a1, b3v[j], uacc[1][j]);
      }
    }
    // SiLU(h) * u  ->  bf16 act tile in LDS
#pragma unroll
    for (int mi = 0; mi < 2; ++mi)
#pragma unroll
      for (int j = 0; j < 3; ++j)
#pragma unroll
        for (int r = 0; r < 8; ++r) {
          const int row = mi * 16 + rbase + r;
          const int col = ncol0 + j * 16;
          float hv = hacc[mi][j][r];
          float uv = uacc[mi][j][r];
          float av = (hv / (1.f + __expf(-hv))) * uv;
          s_act[row * I + col] = f2bf(av);
        }
  }
  __syncthreads();

  // -------- GEMM2: out[32 x 1024] += diag(w) * act * W2
  int   tokr[2][8];
  float wtr[2][8];
#pragma unroll
  for (int mi = 0; mi < 2; ++mi)
#pragma unroll
    for (int r = 0; r < 8; ++r) {
      const int row = mi * 16 + rbase + r;
      tokr[mi][r] = s_tok[row];
      wtr[mi][r]  = s_wt[row];
    }

  for (int p = 0; p < 2; ++p) {                 // wave covers 8 N-tiles, 4 per pass
    v8f oacc[2][4] = {};
    const int nt0   = wv * 8 + p * 4;
    const int ncol0 = nt0 * 16 + n;

#pragma unroll 1
    for (int ks = 0; ks < KBLK_I; ++ks) {
      const int k0 = ks * 32 + kb;
      v16bf a0 = loadA(s_act + (0 * 16 + n) * I, k0);   // LDS: ds_load_b128
      v16bf a1 = loadA(s_act + (1 * 16 + n) * I, k0);
      v16bf b2v[4];
#pragma unroll
      for (int j = 0; j < 4; ++j) {
        const unsigned short* b2p = W2e + ((size_t)(nt0 + j) * KBLK_I + ks) * 512;
        b2v[j] = loadB_swz(b2p, l);
        __builtin_prefetch(b2p + 512, 0, 1);
      }
#pragma unroll
      for (int j = 0; j < 4; ++j) {
        oacc[0][j] = wmma_bf16(a0, b2v[j], oacc[0][j]);
        oacc[1][j] = wmma_bf16(a1, b2v[j], oacc[1][j]);
      }
    }
#pragma unroll
    for (int mi = 0; mi < 2; ++mi)
#pragma unroll
      for (int j = 0; j < 4; ++j)
#pragma unroll
        for (int r = 0; r < 8; ++r) {
          const int nc = ncol0 + j * 16;
          atomicAdd(&out[(size_t)tokr[mi][r] * H + nc], oacc[mi][j][r] * wtr[mi][r]);
        }
  }
}

// ---------------------------------------------------------------- launcher
extern "C" void kernel_launch(void* const* d_in, const int* in_sizes, int n_in,
                              void* d_out, int out_size, void* d_ws, size_t ws_size,
                              hipStream_t stream) {
  const float* x      = (const float*)d_in[0];   // [T,H]
  const float* gate_w = (const float*)d_in[1];   // [H,E]
  const float* w1     = (const float*)d_in[2];   // [E,H,I]
  const float* w3     = (const float*)d_in[3];   // [E,H,I]
  const float* w2     = (const float*)d_in[4];   // [E,I,H]
  float* out = (float*)d_out;

  // workspace carve-up (~85 MB)
  char* w = (char*)d_ws;
  unsigned short* xb  = (unsigned short*)w;  w += (size_t)T * H * 2;
  unsigned short* w1b = (unsigned short*)w;  w += (size_t)E * H * I * 2;
  unsigned short* w3b = (unsigned short*)w;  w += (size_t)E * H * I * 2;
  unsigned short* w2b = (unsigned short*)w;  w += (size_t)E * I * H * 2;
  int*   topk_idx     = (int*)w;             w += (size_t)T * TOPK * 4;
  float* topk_w       = (float*)w;           w += (size_t)T * TOPK * 4;
  int*   token_list   = (int*)w;             w += (size_t)T * TOPK * 4;
  float* weight_list  = (float*)w;           w += (size_t)T * TOPK * 4;
  int*   counts       = (int*)w;             w += 64;
  int*   offsets      = (int*)w;             w += 64;
  int*   cursor       = (int*)w;             w += 64;

  k_zero  <<<(T * H / 4 + 255) / 256, 256, 0, stream>>>(out, counts);
  k_cvt_x <<<(T * H / 4 + 255) / 256, 256, 0, stream>>>(x, xb);

  // swizzle all three weight tensors into WMMA B-operand layout (bf16)
  {
    int thr1 = E * KBLK_H * NBLK_I * 32;   // w1 / w3: K=H, N=I
    int thr2 = E * KBLK_I * NBLK_H * 32;   // w2:      K=I, N=H
    k_swz_w<<<(thr1 + 255) / 256, 256, 0, stream>>>(w1, w1b, KBLK_H, NBLK_I, I);
    k_swz_w<<<(thr1 + 255) / 256, 256, 0, stream>>>(w3, w3b, KBLK_H, NBLK_I, I);
    k_swz_w<<<(thr2 + 255) / 256, 256, 0, stream>>>(w2, w2b, KBLK_I, NBLK_H, H);
  }

  k_router <<<T / 4, 128, 0, stream>>>(x, gate_w, topk_idx, topk_w, counts);
  k_scan   <<<1, 32, 0, stream>>>(counts, offsets, cursor);
  k_scatter<<<(T + 255) / 256, 256, 0, stream>>>(topk_idx, topk_w, cursor,
                                                 token_list, weight_list);
  k_moe_gemm<<<E * TILES_PER_E, 256, 0, stream>>>(xb, w1b, w3b, w2b, offsets, counts,
                                                  token_list, weight_list, out);
}
